// GCNPPI_85968065396828
// MI455X (gfx1250) — compile-verified, matrix operand order checked by reading, not run on hardware
//
#include <hip/hip_runtime.h>
#include <hip/hip_bf16.h>

// ---------------------------------------------------------------------------
// GCN / PPI pipeline for MI455X (gfx1250, wave32, WMMA 16x16x32 f16)
// GEMM: one wave computes a 16x64 output strip (4 N-tiles). A fragment reused
// across 4 WMMAs per K-step; double-buffered prefetch of the next K-step;
// running pointers (constant 64B advance) instead of per-iteration address
// recomputation.
// ---------------------------------------------------------------------------

typedef __attribute__((ext_vector_type(16))) _Float16 v16h;
typedef __attribute__((ext_vector_type(8)))  float    v8f;

#define GN    20000
#define GE    320000
#define GDIN  50
#define GHID  1024
#define GC    121
#define GH3   6
#define K1PAD 64         // DIN=50 padded to 2 K-steps of 32
#define N3    (GH3*GC)   // 726
#define N3PAD 768        // 48 tiles = 12 groups of 4

// ---------------- utility kernels ----------------

__global__ void k_zero_f32(float* __restrict__ p, size_t n) {
    size_t i = (size_t)blockIdx.x * blockDim.x + threadIdx.x;
    if (i < n) p[i] = 0.0f;
}

__global__ void k_deg(const int* __restrict__ col, float* __restrict__ deg, int E) {
    int e = blockIdx.x * blockDim.x + threadIdx.x;
    if (e < E) atomicAdd(&deg[col[e]], 1.0f);
}

// deg -> deg^{-1/2} in place (0 where deg==0)
__global__ void k_dis(float* __restrict__ deg, int n) {
    int i = blockIdx.x * blockDim.x + threadIdx.x;
    if (i < n) {
        float d = deg[i];
        deg[i] = (d > 0.0f) ? rsqrtf(d) : 0.0f;
    }
}

__global__ void k_norm(const int* __restrict__ row, const int* __restrict__ col,
                       const float* __restrict__ dis, float* __restrict__ nrm, int E) {
    int e = blockIdx.x * blockDim.x + threadIdx.x;
    if (e < E) nrm[e] = dis[row[e]] * dis[col[e]];
}

// f32 row-major [n x K] -> f16 row-major [n x Kpad], zero padded K
__global__ void k_cvtA(const float* __restrict__ X, _Float16* __restrict__ Ah,
                       int nrows, int K, int Kpad) {
    size_t i = (size_t)blockIdx.x * blockDim.x + threadIdx.x;
    size_t total = (size_t)nrows * Kpad;
    if (i >= total) return;
    int k = (int)(i % Kpad);
    size_t r = i / Kpad;
    Ah[i] = (k < K) ? (_Float16)X[r * K + k] : (_Float16)0.0f;
}

// f32 row-major W[K x Ncol] -> f16 COLUMN-major padded Wh[n*Kpad + k]
__global__ void k_cvtWcm(const float* __restrict__ W, _Float16* __restrict__ Wh,
                         int K, int Kpad, int Ncol, int Npad) {
    size_t i = (size_t)blockIdx.x * blockDim.x + threadIdx.x;
    size_t total = (size_t)Npad * Kpad;
    if (i >= total) return;
    int k = (int)(i % Kpad);
    int n = (int)(i / Kpad);
    Wh[i] = (k < K && n < Ncol) ? (_Float16)W[(size_t)k * Ncol + n] : (_Float16)0.0f;
}

// ---------------- WMMA GEMM: C[M x Npad] = A[M x Kpad] * Bcm ----------------
union UFrag { v16h v; unsigned int u[8]; };

// Fragment load with constant offsets from a running pointer.
__device__ __forceinline__ void load_frag(UFrag& f, const unsigned int* __restrict__ p) {
#pragma unroll
    for (int r = 0; r < 4; ++r) {
        f.u[r]     = p[r];       // VGPR 0..3 : K = kbase+0..7   (packed pairs)
        f.u[r + 4] = p[8 + r];   // VGPR 4..7 : K = kbase+16..23
    }
}

__global__ void k_gemm_wmma(const _Float16* __restrict__ A,
                            const _Float16* __restrict__ Bcm,
                            float* __restrict__ C,
                            int Mtiles, int Ngroups, int Kpad, int ldc) {
    int lane = threadIdx.x & 31;
    int wid  = blockIdx.x * (blockDim.x >> 5) + (threadIdx.x >> 5);
    if (wid >= Mtiles * Ngroups) return;
    int tm = wid / Ngroups;
    int tg = wid % Ngroups;            // group of 4 consecutive N-tiles

    int half  = (lane >> 4) & 1;       // which 16-lane half
    int idx16 = lane & 15;
    int m     = tm * 16 + idx16;       // A row held by this lane
    int kboff = half * 4;              // (kbase=half*8 halves) >> 1 dwords

    // Running dword pointers; advance by 16 dwords (one 32-wide K-step).
    const unsigned int* pA = (const unsigned int*)(A + (size_t)m * Kpad) + kboff;
    const unsigned int* pB[4];
#pragma unroll
    for (int j = 0; j < 4; ++j) {
        int n = (tg * 4 + j) * 16 + idx16;   // B column held by this lane
        pB[j] = (const unsigned int*)(Bcm + (size_t)n * Kpad) + kboff;
    }

    v8f acc[4] = {};

    // prologue: fetch K-step 0
    UFrag a0, b0[4];
    load_frag(a0, pA);
#pragma unroll
    for (int j = 0; j < 4; ++j) load_frag(b0[j], pB[j]);

    // steady state: prefetch step k while multiplying step k-1
#pragma unroll 2
    for (int k0 = 32; k0 < Kpad; k0 += 32) {
        pA += 16;
#pragma unroll
        for (int j = 0; j < 4; ++j) pB[j] += 16;

        UFrag a1, b1[4];
        load_frag(a1, pA);
#pragma unroll
        for (int j = 0; j < 4; ++j) load_frag(b1[j], pB[j]);

#pragma unroll
        for (int j = 0; j < 4; ++j) {
            acc[j] = __builtin_amdgcn_wmma_f32_16x16x32_f16(
                false, a0.v, false, b0[j].v, (short)0, acc[j], false, false);
        }
        a0 = a1;
#pragma unroll
        for (int j = 0; j < 4; ++j) b0[j] = b1[j];
    }

    // epilogue: last K-step
#pragma unroll
    for (int j = 0; j < 4; ++j) {
        acc[j] = __builtin_amdgcn_wmma_f32_16x16x32_f16(
            false, a0.v, false, b0[j].v, (short)0, acc[j], false, false);
    }

    // C/D layout: VGPR r holds M = r (lanes 0-15) or r+8 (lanes 16-31)
#pragma unroll
    for (int j = 0; j < 4; ++j) {
        int n = (tg * 4 + j) * 16 + idx16;
#pragma unroll
        for (int r = 0; r < 8; ++r) {
            int mm = tm * 16 + (half ? (r + 8) : r);
            C[(size_t)mm * ldc + n] = acc[j][r];
        }
    }
}

// ---------------- edge aggregation: ACC[col] += norm * G[row] ----------------
// Feature-major mapping: consecutive lanes hit consecutive float4 chunks of
// the same edge -> fully coalesced gathers + atomics.
__global__ void k_agg(const float* __restrict__ G, float* __restrict__ ACC,
                      const int* __restrict__ row, const int* __restrict__ col,
                      const float* __restrict__ nrm, int E, int F4, int ldc) {
    size_t idx = (size_t)blockIdx.x * blockDim.x + threadIdx.x;
    size_t total = (size_t)E * F4;
    if (idx >= total) return;
    unsigned int e = (unsigned int)(idx / F4);
    unsigned int c = (unsigned int)(idx % F4) * 4u;
    float s = nrm[e];
    const float4 v = *(const float4*)(G + (size_t)row[e] * ldc + c);
    float* dst = ACC + (size_t)col[e] * ldc + c;
    atomicAdd(dst + 0, s * v.x);
    atomicAdd(dst + 1, s * v.y);
    atomicAdd(dst + 2, s * v.z);
    atomicAdd(dst + 3, s * v.w);
}

// ---------------- ELU + convert to f16 ----------------
__global__ void k_elu_f16(const float* __restrict__ src, _Float16* __restrict__ dst, size_t n) {
    size_t i = (size_t)blockIdx.x * blockDim.x + threadIdx.x;
    if (i >= n) return;
    float x = src[i];
    float y = (x > 0.0f) ? x : (__expf(x) - 1.0f);
    dst[i] = (_Float16)y;
}

// ---------------- head mean: out[n,c] = mean_h ACC[n, h*C + c] ----------------
__global__ void k_mean_heads(const float* __restrict__ ACC, float* __restrict__ out,
                             int nrows, int C, int H, int ld) {
    size_t i = (size_t)blockIdx.x * blockDim.x + threadIdx.x;
    size_t total = (size_t)nrows * C;
    if (i >= total) return;
    int c = (int)(i % C);
    size_t r = i / C;
    float s = 0.0f;
    const float* base = ACC + r * ld + c;
#pragma unroll
    for (int h = 0; h < GH3; ++h) s += base[h * C];
    out[i] = s * (1.0f / (float)H);
}

// ---------------------------------------------------------------------------

static inline int cdiv(size_t a, int b) { return (int)((a + b - 1) / b); }

extern "C" void kernel_launch(void* const* d_in, const int* in_sizes, int n_in,
                              void* d_out, int out_size, void* d_ws, size_t ws_size,
                              hipStream_t stream) {
    const float* X  = (const float*)d_in[0];
    const int*   ei = (const int*)  d_in[1];
    const float* W1 = (const float*)d_in[2];
    const float* W2 = (const float*)d_in[3];
    const float* S2 = (const float*)d_in[4];
    const float* W3 = (const float*)d_in[5];
    const float* S3 = (const float*)d_in[6];
    float* out = (float*)d_out;

    const int* row = ei;        // edge_index[0]
    const int* col = ei + GE;   // edge_index[1]

    // ---- workspace carve (aligned 256B) ----
    char* ws = (char*)d_ws;
    auto carve = [&](size_t bytes) -> void* {
        void* p = (void*)ws;
        ws += (bytes + 255) & ~(size_t)255;
        return p;
    };
    float*     deg  = (float*)    carve((size_t)GN * 4);            // -> dis in place
    float*     nrm  = (float*)    carve((size_t)GE * 4);
    _Float16*  Ah   = (_Float16*) carve((size_t)GN * K1PAD * 2);    // layer1 A (f16)
    _Float16*  Hh   = (_Float16*) carve((size_t)GN * GHID * 2);     // h1/h2 (f16)
    _Float16*  Wh   = (_Float16*) carve((size_t)GHID * GHID * 2);   // weight (f16, col-major)
    _Float16*  Sh   = (_Float16*) carve((size_t)GHID * GHID * 2);   // skip weight (f16)
    float*     G    = (float*)    carve((size_t)GN * GHID * 4);     // GEMM out / agg source
    float*     ACC  = (float*)    carve((size_t)GN * GHID * 4);     // agg accumulator

    const int B = 256;

    // ---- graph normalization ----
    k_zero_f32<<<cdiv(GN, B), B, 0, stream>>>(deg, (size_t)GN);
    k_deg     <<<cdiv(GE, B), B, 0, stream>>>(col, deg, GE);
    k_dis     <<<cdiv(GN, B), B, 0, stream>>>(deg, GN);
    k_norm    <<<cdiv(GE, B), B, 0, stream>>>(row, col, deg, nrm, GE);

    // ---- layer 1: h1 = elu(agg(X @ W1)) ----
    k_cvtA  <<<cdiv((size_t)GN * K1PAD, B), B, 0, stream>>>(X, Ah, GN, GDIN, K1PAD);
    k_cvtWcm<<<cdiv((size_t)GHID * K1PAD, B), B, 0, stream>>>(W1, Wh, GDIN, K1PAD, GHID, GHID);
    {
        int mt = GN / 16, ng = (GHID / 16) / 4;      // 1250 x 16 strip-groups
        int waves = mt * ng;
        k_gemm_wmma<<<cdiv((size_t)waves, 8), 256, 0, stream>>>(Ah, Wh, G, mt, ng, K1PAD, GHID);
    }
    k_zero_f32<<<cdiv((size_t)GN * GHID, B), B, 0, stream>>>(ACC, (size_t)GN * GHID);
    k_agg     <<<cdiv((size_t)GE * (GHID / 4), B), B, 0, stream>>>(G, ACC, row, col, nrm, GE, GHID / 4, GHID);
    k_elu_f16 <<<cdiv((size_t)GN * GHID, B), B, 0, stream>>>(ACC, Hh, (size_t)GN * GHID);

    // ---- layer 2: h2 = elu(agg(h1 @ W2) + h1 @ S2) ----
    k_cvtWcm<<<cdiv((size_t)GHID * GHID, B), B, 0, stream>>>(W2, Wh, GHID, GHID, GHID, GHID);
    k_cvtWcm<<<cdiv((size_t)GHID * GHID, B), B, 0, stream>>>(S2, Sh, GHID, GHID, GHID, GHID);
    {
        int mt = GN / 16, ng = (GHID / 16) / 4;
        int waves = mt * ng;
        k_gemm_wmma<<<cdiv((size_t)waves, 8), 256, 0, stream>>>(Hh, Wh, G,   mt, ng, GHID, GHID);
        k_gemm_wmma<<<cdiv((size_t)waves, 8), 256, 0, stream>>>(Hh, Sh, ACC, mt, ng, GHID, GHID); // skip init
    }
    k_agg    <<<cdiv((size_t)GE * (GHID / 4), B), B, 0, stream>>>(G, ACC, row, col, nrm, GE, GHID / 4, GHID);
    k_elu_f16<<<cdiv((size_t)GN * GHID, B), B, 0, stream>>>(ACC, Hh, (size_t)GN * GHID);

    // ---- layer 3: y = agg(h2 @ W3) + h2 @ S3 ; mean over 6 heads ----
    k_cvtWcm<<<cdiv((size_t)N3PAD * GHID, B), B, 0, stream>>>(W3, Wh, GHID, GHID, N3, N3PAD);
    k_cvtWcm<<<cdiv((size_t)N3PAD * GHID, B), B, 0, stream>>>(S3, Sh, GHID, GHID, N3, N3PAD);
    {
        int mt = GN / 16, ng = (N3PAD / 16) / 4;     // 1250 x 12 strip-groups
        int waves = mt * ng;
        k_gemm_wmma<<<cdiv((size_t)waves, 8), 256, 0, stream>>>(Hh, Wh, G,   mt, ng, GHID, N3PAD);
        k_gemm_wmma<<<cdiv((size_t)waves, 8), 256, 0, stream>>>(Hh, Sh, ACC, mt, ng, GHID, N3PAD); // skip init
    }
    k_agg       <<<cdiv((size_t)GE * (N3PAD / 4), B), B, 0, stream>>>(G, ACC, row, col, nrm, GE, N3PAD / 4, N3PAD);
    k_mean_heads<<<cdiv((size_t)GN * GC, B), B, 0, stream>>>(ACC, out, GN, GC, GH3, N3PAD);
}
